// HBondNetworkAnalyzer_79534204387845
// MI455X (gfx1250) — matrix operand order repaired and checked by reading, not compiled, not run on hardware
//
#include <hip/hip_runtime.h>
#include <cmath>

#define Bq 2
#define Lq 512
#define Fq 256
#define Hq 128
#define H2 64
#define MTILES 8
#define MGROUPS (Lq / (16 * MTILES))   // 4

typedef __attribute__((ext_vector_type(16))) __bf16 v16bf;
typedef __attribute__((ext_vector_type(8)))  __bf16 v8bf;
typedef __attribute__((ext_vector_type(8)))  float  v8f;

union V16BF { v16bf v; v8bf h[2]; };

// ---------------------------------------------------------------------------
// Kernel 0: zero the two energy accumulators in d_out (harness poisons d_out)
// ---------------------------------------------------------------------------
__global__ void init_energy(float* outE) {
    if (threadIdx.x < Bq) outE[threadIdx.x] = 0.0f;
}

// ---------------------------------------------------------------------------
// Kernel 1: per-residue row projections  dp = feat@Wd (+bias), ap = feat@Wa
// for both paths.  O(L*F*H), tiny vs the pair loop.
// ---------------------------------------------------------------------------
__global__ void precompute_rows(const float* __restrict__ feat,
                                const float* __restrict__ W1,
                                const float* __restrict__ b1,
                                const float* __restrict__ V1,
                                const float* __restrict__ c1,
                                float* __restrict__ dp1, float* __restrict__ ap1,
                                float* __restrict__ dpV, float* __restrict__ apV) {
    int i  = blockIdx.x * blockDim.x + threadIdx.x;   // over B*L*H
    int h  = i & (Hq - 1);
    int bl = i >> 7;
    const float* frow = feat + (size_t)bl * Fq;
    float a0 = 0.f, a1 = 0.f, a2 = 0.f, a3 = 0.f;
    for (int f = 0; f < Fq; ++f) {
        float x = frow[f];
        a0 += x * W1[f * Hq + h];
        a1 += x * W1[(Fq + f) * Hq + h];
        a2 += x * V1[f * Hq + h];
        a3 += x * V1[(Fq + f) * Hq + h];
    }
    dp1[i] = a0 + b1[h];
    ap1[i] = a1;
    dpV[i] = a2 + c1[h];
    apV[i] = a3;
}

// ---------------------------------------------------------------------------
// Kernel 2: fold BN into W2, transpose weights to n-major bf16, build b2'
// ---------------------------------------------------------------------------
__global__ void precompute_weights(const float* __restrict__ W2, const float* __restrict__ b2,
                                   const float* __restrict__ gamma, const float* __restrict__ beta,
                                   const float* __restrict__ mean, const float* __restrict__ var,
                                   const float* __restrict__ V2,
                                   __bf16* __restrict__ W2t, __bf16* __restrict__ V2t,
                                   float* __restrict__ b2p) {
    int n = blockIdx.x;    // 0..63
    int k = threadIdx.x;   // 0..127
    float s = gamma[k] * rsqrtf(var[k] + 1e-5f);
    float t = beta[k] - mean[k] * s;
    float w = W2[k * H2 + n];
    W2t[n * Hq + k] = (__bf16)(w * s);
    V2t[n * Hq + k] = (__bf16)V2[k * H2 + n];
    __shared__ float red[128];
    red[k] = t * w;
    __syncthreads();
    for (int off = 64; off > 0; off >>= 1) {
        if (k < off) red[k] += red[k + off];
        __syncthreads();
    }
    if (k == 0) b2p[n] = b2[n] + red[0];
}

// ---------------------------------------------------------------------------
// Kernel 3: pair MLPs via WMMA.  Block = (b, l, m-group of 128 m's),
// 8 waves = 2 paths x 4 N-tiles, loop over 8 m-tiles of 16 pairs.
// Epilogue stores w3-prescaled relu(h2) so the 64->1 head is a pure sum,
// reduced by a two-stage parallel tree (128 threads, then 16).
// ---------------------------------------------------------------------------
__global__ __launch_bounds__(256)
void pair_kernel(const float* __restrict__ structures,
                 const int* __restrict__ donor, const int* __restrict__ accept,
                 const float* __restrict__ W1, const float* __restrict__ V1,
                 const float* __restrict__ dp1, const float* __restrict__ ap1,
                 const float* __restrict__ dpV, const float* __restrict__ apV,
                 const __bf16* __restrict__ W2t, const __bf16* __restrict__ V2t,
                 const float* __restrict__ b2p, const float* __restrict__ c2,
                 const float* __restrict__ W3, const float* __restrict__ b3,
                 const float* __restrict__ V3, const float* __restrict__ c3,
                 float* __restrict__ outA, float* __restrict__ outS, float* __restrict__ outE) {
    const int t    = threadIdx.x;
    const int lane = t & 31;
    const int wave = t >> 5;
    const int path  = wave >> 2;   // 0 = adjacency MLP, 1 = strength MLP
    const int ntile = wave & 3;
    const int n0    = ntile * 16;
    const int mgrp = blockIdx.x;
    const int l    = blockIdx.y;
    const int b    = blockIdx.z;

    __shared__ __bf16 As[2][16][136];       // relu(z) in bf16, row-padded
    __shared__ float dps[2][128];
    __shared__ float wg[2][3][128];
    __shared__ float pwb[2][16][68];        // w3-prescaled relu(h2), padded
    __shared__ float partial[2][16][4];     // stage-1 head partials
    __shared__ float dfp[16], sepp[16], valp[16];
    __shared__ float cb2[64], cc2[64], cw3[64], cv3[64];
    __shared__ float sl[3];
    __shared__ float sb3, sc3, donorf, eacc;

    // ---- one-time block init ----
    if (t < 128) {
        dps[0][t] = dp1[((size_t)b * Lq + l) * Hq + t];
        dps[1][t] = dpV[((size_t)b * Lq + l) * Hq + t];
        wg[0][0][t] = W1[(2 * Fq + 0) * Hq + t];
        wg[0][1][t] = W1[(2 * Fq + 1) * Hq + t];
        wg[0][2][t] = W1[(2 * Fq + 2) * Hq + t];
        wg[1][0][t] = V1[(2 * Fq + 0) * Hq + t];
        wg[1][1][t] = V1[(2 * Fq + 1) * Hq + t];
        wg[1][2][t] = V1[(2 * Fq + 2) * Hq + t];
    } else {
        int u = t - 128;
        if (u < 64) { cb2[u] = b2p[u]; cc2[u] = c2[u]; cw3[u] = W3[u]; cv3[u] = V3[u]; }
        else if (u == 64) {
            sb3 = b3[0]; sc3 = c3[0];
            donorf = (float)donor[b * Lq + l];
            eacc = 0.f;
            sl[0] = structures[((size_t)b * Lq + l) * 3 + 0];
            sl[1] = structures[((size_t)b * Lq + l) * 3 + 1];
            sl[2] = structures[((size_t)b * Lq + l) * 3 + 2];
        }
    }

    // ---- per-wave B fragments (stay in registers for all m-tiles) ----
    // B layout chosen here (A matches): lane holds column n0+(lane&15); K window base is
    // (lane&16)?16:0 within each 32-K step; 16 contiguous K per fragment.
    const __bf16* Wt = path ? V2t : W2t;
    V16BF bfrag[4];
    {
        int n   = n0 + (lane & 15);
        int kb0 = (lane & 16) ? 16 : 0;
        for (int kk = 0; kk < 4; ++kk) {
            const v8bf* p0 = (const v8bf*)(Wt + n * Hq + 32 * kk + kb0);
            bfrag[kk].h[0] = p0[0];
            bfrag[kk].h[1] = p0[1];
        }
    }
    __syncthreads();

    // per-lane epilogue constants (uniform across m-tiles)
    const int   ncol  = n0 + (lane & 15);
    const float bias2 = path ? cc2[ncol] : cb2[ncol];
    const float w3n   = path ? cv3[ncol] : cw3[ncol];
    const int   mbase = (lane & 16) ? 8 : 0;

    for (int mt = 0; mt < MTILES; ++mt) {
        const int m0 = mgrp * (16 * MTILES) + mt * 16;

        // --- per-pair geometry scalars ---
        if (t < 16) {
            int m = m0 + t;
            float dx = sl[0] - structures[((size_t)b * Lq + m) * 3 + 0];
            float dy = sl[1] - structures[((size_t)b * Lq + m) * 3 + 1];
            float dz = sl[2] - structures[((size_t)b * Lq + m) * 3 + 2];
            float dist = sqrtf(dx * dx + dy * dy + dz * dz);
            dfp[t] = __expf(dist * (-1.0f / 3.5f));
            float sep = fabsf((float)(l - m));
            sepp[t] = sep * 0.1f;
            valp[t] = (donorf != 0.f && accept[b * Lq + m] != 0 && sep >= 2.f) ? 1.f : 0.f;
        }
        __syncthreads();

        // --- build A = relu(dp[l] + ap[m] + geom@Wg + bias) in bf16 ---
        {
            int p = t >> 4;          // pair row 0..15
            int q = t & 15;          // 8-wide k chunk
            int m = m0 + p;
            float df = dfp[p], sf = sepp[p];
            size_t abase = ((size_t)b * Lq + m) * Hq + q * 8;
            for (int j = 0; j < 8; ++j) {
                int k = q * 8 + j;
                float z0 = dps[0][k] + ap1[abase + j] + df * wg[0][0][k] + wg[0][1][k] + sf * wg[0][2][k];
                As[0][p][k] = (__bf16)fmaxf(z0, 0.f);
                float z1 = dps[1][k] + apV[abase + j] + df * wg[1][0][k] + wg[1][1][k] + sf * wg[1][2][k];
                As[1][p][k] = (__bf16)fmaxf(z1, 0.f);
            }
        }
        __syncthreads();

        // --- WMMA: (16 pairs x 128) @ (128 x 16 cols), K chained in 4 steps.
        //     All 8 A-fragment ds_load_b128 issue first, then WMMAs chain. ---
        v8f acc = {};
        {
            int p = lane & 15;
            int koffh = (lane & 16) ? 16 : 0;  // A: lane-half K offset
            const __bf16* arow = &As[path][p][0];
            V16BF a[4];
#pragma unroll
            for (int kk = 0; kk < 4; ++kk) {
                a[kk].h[0] = *(const v8bf*)(arow + 32 * kk + koffh);        // K = base..base+7
                a[kk].h[1] = *(const v8bf*)(arow + 32 * kk + koffh + 8);    // K = base+8..+15
            }
#pragma unroll
            for (int kk = 0; kk < 4; ++kk)
                acc = __builtin_amdgcn_wmma_f32_16x16x32_bf16(
                        false, a[kk].v, false, bfrag[kk].v, (short)0, acc, false, false);
        }

        // --- epilogue: pw = relu(acc + b2') * w3[n] -> LDS (head is now a sum) ---
        {
            int c = n0 + (lane & 15);
#pragma unroll
            for (int v = 0; v < 8; ++v)
                pwb[path][mbase + v][c] = fmaxf(acc[v] + bias2, 0.f) * w3n;
        }
        __syncthreads();

        // --- head stage 1: 128 threads, each sums 16 contiguous floats ---
        if (t < 128) {
            int pt = t >> 6, row = (t >> 2) & 15, c = t & 3;
            const float* src = &pwb[pt][row][c * 16];
            float s = 0.f;
#pragma unroll
            for (int j = 0; j < 16; ++j) s += src[j];
            partial[pt][row][c] = s;
        }
        __syncthreads();

        // --- head stage 2 + outputs: sigmoid(x)>0.5 <=> x>0 for adjacency ---
        if (t < 16) {
            int m = m0 + t;
            float z0 = partial[0][t][0] + partial[0][t][1] + partial[0][t][2] + partial[0][t][3] + sb3;
            float z1 = partial[1][t][0] + partial[1][t][1] + partial[1][t][2] + partial[1][t][3] + sc3;
            float adj = (z0 > 0.f) ? valp[t] : 0.f;
            float strength = 1.f / (1.f + __expf(-z1));
            float sv = adj * strength;
            size_t o = ((size_t)b * Lq + l) * Lq + m;
            outA[o] = adj;
            outS[o] = sv;
            atomicAdd(&eacc, sv);
        }
        __syncthreads();
    }
    if (t == 0) atomicAdd(&outE[b], -5.0f * eacc);
}

// ---------------------------------------------------------------------------
extern "C" void kernel_launch(void* const* d_in, const int* in_sizes, int n_in,
                              void* d_out, int out_size, void* d_ws, size_t ws_size,
                              hipStream_t stream) {
    (void)in_sizes; (void)n_in; (void)out_size; (void)ws_size;
    const float* feat       = (const float*)d_in[0];
    const float* structures = (const float*)d_in[1];
    const int*   donor      = (const int*)d_in[2];
    const int*   accept     = (const int*)d_in[3];
    const float* W1 = (const float*)d_in[4];
    const float* b1 = (const float*)d_in[5];
    const float* g  = (const float*)d_in[6];
    const float* be = (const float*)d_in[7];
    const float* mu = (const float*)d_in[8];
    const float* va = (const float*)d_in[9];
    const float* W2 = (const float*)d_in[10];
    const float* b2 = (const float*)d_in[11];
    const float* W3 = (const float*)d_in[12];
    const float* b3 = (const float*)d_in[13];
    const float* V1 = (const float*)d_in[14];
    const float* c1 = (const float*)d_in[15];
    const float* V2 = (const float*)d_in[16];
    const float* c2 = (const float*)d_in[17];
    const float* V3 = (const float*)d_in[18];
    const float* c3 = (const float*)d_in[19];

    char* ws = (char*)d_ws;
    float*  dp1 = (float*)(ws + 0);
    float*  ap1 = (float*)(ws + 524288);
    float*  dpV = (float*)(ws + 1048576);
    float*  apV = (float*)(ws + 1572864);
    __bf16* W2t = (__bf16*)(ws + 2097152);
    __bf16* V2t = (__bf16*)(ws + 2097152 + 16384);
    float*  b2p = (float*)(ws + 2097152 + 32768);

    float* outA = (float*)d_out;
    float* outS = outA + (size_t)Bq * Lq * Lq;
    float* outE = outS + (size_t)Bq * Lq * Lq;

    init_energy<<<1, 32, 0, stream>>>(outE);
    precompute_rows<<<(Bq * Lq * Hq) / 256, 256, 0, stream>>>(feat, W1, b1, V1, c1,
                                                              dp1, ap1, dpV, apV);
    precompute_weights<<<64, 128, 0, stream>>>(W2, b2, g, be, mu, va, V2, W2t, V2t, b2p);

    dim3 grid(MGROUPS, Lq, Bq);
    pair_kernel<<<grid, 256, 0, stream>>>(structures, donor, accept, W1, V1,
                                          dp1, ap1, dpV, apV, W2t, V2t, b2p, c2,
                                          W3, b3, V3, c3, outA, outS, outE);
}
